// GlobalSelfAttention_28698971472611
// MI455X (gfx1250) — matrix operand order repaired and checked by reading, not compiled, not run on hardware
//
#include <hip/hip_runtime.h>
#include <hip/hip_bf16.h>

typedef __attribute__((ext_vector_type(16))) _Float16 v16h;
typedef __attribute__((ext_vector_type(8)))  _Float16 v8h;
typedef __attribute__((ext_vector_type(8)))  float    v8f;

#define S_LEN 2048
#define D_DIM 1024
#define H_NUM 16
#define K_DIM 64
#define EPS   1e-3f

// ---------------------------------------------------------------------------
// WMMA helpers (gfx1250 wave32, 16x16x32 f16 -> f32)
// ---------------------------------------------------------------------------
__device__ inline v8f wmma_f16(v16h a, v16h b, v8f c) {
    // (neg_a, A, neg_b, B, c_mod, C, reuse_a, reuse_b)
    return __builtin_amdgcn_wmma_f32_16x16x32_f16(false, a, false, b,
                                                  (short)0, c, false, false);
}

// A fragment: 16x32 f16 tile, row-major memory, stride lda (halves).
// Lane l: row = l%16, kbase = 8*(l/16); elems 0..7 = K[kbase..kbase+7],
// elems 8..15 = K[kbase+16..kbase+23]  (ISA 7.12.2, 16-bit A 16x32).
__device__ inline v16h load_a_frag(const _Float16* __restrict__ base, int lda) {
    int lane = threadIdx.x & 31;
    int row  = lane & 15;
    int kb   = (lane >> 4) << 3;
    const _Float16* p = base + row * lda + kb;
    union { v16h v; v8h h[2]; } u;
    u.h[0] = *(const v8h*)(p);
    u.h[1] = *(const v8h*)(p + 16);
    return u.v;
}

// B fragment: 32x16 f16 tile where B[k][n] = baseT[n*ldt + k] (transposed
// storage: each output column is contiguous along K). Lane l: col = l%16,
// kbase = 8*(l/16); elems 0..7 = K[kbase..kbase+7], elems 8..15 = K[kbase+16..kbase+23].
__device__ inline v16h load_b_frag_T(const _Float16* __restrict__ baseT, int ldt) {
    int lane = threadIdx.x & 31;
    int n    = lane & 15;
    int kb   = (lane >> 4) << 3;
    const _Float16* p = baseT + n * ldt + kb;
    union { v16h v; v8h h[2]; } u;
    u.h[0] = *(const v8h*)(p);
    u.h[1] = *(const v8h*)(p + 16);
    return u.v;
}

// ---------------------------------------------------------------------------
// Pack / transpose-pack fp32 -> f16
// ---------------------------------------------------------------------------
__global__ void pack_f16_kernel(const float* __restrict__ in,
                                _Float16* __restrict__ out, int n) {
    int i = blockIdx.x * blockDim.x + threadIdx.x;
    if (i < n) out[i] = (_Float16)in[i];
}

// in: [rows][cols] f32 -> out: [cols][rows] f16
__global__ void transpose_pack_kernel(const float* __restrict__ in,
                                      _Float16* __restrict__ out,
                                      int rows, int cols) {
    int i = blockIdx.x * blockDim.x + threadIdx.x;
    if (i < rows * cols) {
        int r = i / cols, c = i - r * cols;
        out[c * rows + r] = (_Float16)in[i];
    }
}

// ---------------------------------------------------------------------------
// QKV projection GEMM:  C[m][n] = sum_d x[m][d] * W[d][n] + bias[n]
// A = x_f16 [8192][1024]; BT = W^T f16 [1024][1024].
// transpose_out==0: out[((b*H+h)*S+s)*64 + kk]   (q, k; scale folds 1/sqrt(K))
// transpose_out==1: out[((b*H+h)*64+kk)*S + s]   (v stored transposed)
// Grid: (M/16, N/512), block 256 (8 waves); wave w owns n-slab of 64.
// ---------------------------------------------------------------------------
__global__ __launch_bounds__(256) void qkv_gemm_kernel(
    const _Float16* __restrict__ A, const _Float16* __restrict__ BT,
    const float* __restrict__ bias, _Float16* __restrict__ out,
    float scale, int transpose_out) {
    const int Kd = D_DIM;
    int m0   = blockIdx.x * 16;
    int wave = threadIdx.x >> 5;
    int n0   = blockIdx.y * 512 + wave * 64;

    v8f acc[4] = {};
    for (int kk = 0; kk < Kd; kk += 32) {
        v16h a = load_a_frag(A + m0 * Kd + kk, Kd);
#pragma unroll
        for (int j = 0; j < 4; ++j) {
            v16h b = load_b_frag_T(BT + (n0 + 16 * j) * Kd + kk, Kd);
            acc[j] = wmma_f16(a, b, acc[j]);
        }
    }

    int lane = threadIdx.x & 31;
    int colg = lane & 15;
    int rowg = (lane >> 4) << 3;
#pragma unroll
    for (int j = 0; j < 4; ++j) {
        int ncol = n0 + 16 * j + colg;
        int h = ncol >> 6, kk2 = ncol & 63;
        float bv = bias[ncol];
#pragma unroll
        for (int r = 0; r < 8; ++r) {
            int m = m0 + rowg + r;
            int bidx = m >> 11;            // / S_LEN
            int s    = m & (S_LEN - 1);
            float val = (acc[j][r] + bv) * scale;
            if (!transpose_out)
                out[(((bidx * H_NUM + h) * S_LEN + s) << 6) + kk2] = (_Float16)val;
            else
                out[(((bidx * H_NUM + h) << 6) + kk2) * S_LEN + s] = (_Float16)val;
        }
    }
}

// ---------------------------------------------------------------------------
// Flash attention: one wave per (b,h,16-query tile). q pre-scaled by 1/8.
// q,k: [B][H][S][64]; vT: [B][H][64][S]; ctx: [B*S][1024] (col = h*64+kk)
// ---------------------------------------------------------------------------
__global__ __launch_bounds__(32) void attn_kernel(
    const _Float16* __restrict__ q, const _Float16* __restrict__ k,
    const _Float16* __restrict__ vT, _Float16* __restrict__ ctx) {
    __shared__ __align__(16) _Float16 P[16][32];

    int bh = blockIdx.y;                    // b*H + h
    int q0 = blockIdx.x * 16;
    const _Float16* qb = q  + (size_t)bh * S_LEN * K_DIM;
    const _Float16* kb = k  + (size_t)bh * S_LEN * K_DIM;
    const _Float16* vb = vT + (size_t)bh * K_DIM * S_LEN;

    int lane = threadIdx.x & 31;
    int colg = lane & 15;
    int rowg = (lane >> 4) << 3;

    v16h qf0 = load_a_frag(qb + q0 * K_DIM + 0, K_DIM);
    v16h qf1 = load_a_frag(qb + q0 * K_DIM + 32, K_DIM);

    v8f ctx_acc[4] = {};
    float mrow[8], lrow[8];
#pragma unroll
    for (int r = 0; r < 8; ++r) { mrow[r] = -1e30f; lrow[r] = 0.f; }

    for (int s0 = 0; s0 < S_LEN; s0 += 32) {
        float p0[8], p1[8];
#pragma unroll
        for (int kt = 0; kt < 2; ++kt) {
            v8f sc = {};
            v16h b0 = load_b_frag_T(kb + (s0 + 16 * kt) * K_DIM + 0,  K_DIM);
            sc = wmma_f16(qf0, b0, sc);
            v16h b1 = load_b_frag_T(kb + (s0 + 16 * kt) * K_DIM + 32, K_DIM);
            sc = wmma_f16(qf1, b1, sc);
#pragma unroll
            for (int r = 0; r < 8; ++r) {
                if (kt == 0) p0[r] = sc[r]; else p1[r] = sc[r];
            }
        }
        // online softmax row update (rows live in fixed acc slots per 16-lane group)
        float alpha[8];
#pragma unroll
        for (int r = 0; r < 8; ++r) {
            float tmax = fmaxf(p0[r], p1[r]);
#pragma unroll
            for (int mask = 8; mask >= 1; mask >>= 1)
                tmax = fmaxf(tmax, __shfl_xor(tmax, mask, 16));
            float newm = fmaxf(mrow[r], tmax);
            alpha[r] = __expf(mrow[r] - newm);
            p0[r] = __expf(p0[r] - newm);
            p1[r] = __expf(p1[r] - newm);
            float rsum = p0[r] + p1[r];
#pragma unroll
            for (int mask = 8; mask >= 1; mask >>= 1)
                rsum += __shfl_xor(rsum, mask, 16);
            lrow[r] = lrow[r] * alpha[r] + rsum;
            mrow[r] = newm;
        }
#pragma unroll
        for (int j = 0; j < 4; ++j)
#pragma unroll
            for (int r = 0; r < 8; ++r) ctx_acc[j][r] *= alpha[r];

        // reshape P: C layout -> A layout via LDS
#pragma unroll
        for (int r = 0; r < 8; ++r) {
            P[rowg + r][colg]      = (_Float16)p0[r];
            P[rowg + r][16 + colg] = (_Float16)p1[r];
        }
        __syncthreads();
        v16h pf = load_a_frag(&P[0][0], 32);
        __syncthreads();

#pragma unroll
        for (int j = 0; j < 4; ++j) {
            v16h vf = load_b_frag_T(vb + (16 * j) * S_LEN + s0, S_LEN);
            ctx_acc[j] = wmma_f16(pf, vf, ctx_acc[j]);
        }
    }

    int bidx = bh >> 4, h = bh & 15;
#pragma unroll
    for (int j = 0; j < 4; ++j)
#pragma unroll
        for (int r = 0; r < 8; ++r) {
            float val = ctx_acc[j][r] / lrow[r];
            int m = bidx * S_LEN + q0 + rowg + r;
            ctx[(size_t)m * D_DIM + h * 64 + 16 * j + colg] = (_Float16)val;
        }
}

// ---------------------------------------------------------------------------
// Output projection + bias + residual + LayerNorm.
// ctx [8192][1024] f16, WoT [1024 d][1024 hk] f16, x fp32 residual.
// Grid 512, block 256: 16 rows per WG; wave w owns 128 cols; y staged in LDS.
// ---------------------------------------------------------------------------
__global__ __launch_bounds__(256) void out_proj_ln_kernel(
    const _Float16* __restrict__ ctx, const _Float16* __restrict__ WoT,
    const float* __restrict__ bo, const float* __restrict__ x,
    const float* __restrict__ gamma, const float* __restrict__ beta,
    float* __restrict__ out) {
    __shared__ __align__(16) float Y[16][D_DIM];   // 64 KB

    int m0   = blockIdx.x * 16;
    int wave = threadIdx.x >> 5;
    int lane = threadIdx.x & 31;
    int n0   = wave * 128;

    v8f acc[8] = {};
    for (int kk = 0; kk < D_DIM; kk += 32) {
        v16h a = load_a_frag(ctx + (size_t)m0 * D_DIM + kk, D_DIM);
#pragma unroll
        for (int j = 0; j < 8; ++j) {
            v16h b = load_b_frag_T(WoT + (n0 + 16 * j) * D_DIM + kk, D_DIM);
            acc[j] = wmma_f16(a, b, acc[j]);
        }
    }

    int colg = lane & 15;
    int rowg = (lane >> 4) << 3;
#pragma unroll
    for (int j = 0; j < 8; ++j) {
        int d = n0 + 16 * j + colg;
        float bv = bo[d];
#pragma unroll
        for (int r = 0; r < 8; ++r) {
            int m = m0 + rowg + r;
            Y[rowg + r][d] = acc[j][r] + bv + x[(size_t)m * D_DIM + d];
        }
    }
    __syncthreads();

    // LayerNorm: wave w normalizes rows 2w and 2w+1
#pragma unroll
    for (int rr = 0; rr < 2; ++rr) {
        int r = wave * 2 + rr;
        float s = 0.f, sq = 0.f;
#pragma unroll
        for (int i = 0; i < 32; ++i) {
            float yv = Y[r][lane + 32 * i];
            s += yv; sq += yv * yv;
        }
#pragma unroll
        for (int mask = 16; mask >= 1; mask >>= 1) {
            s  += __shfl_xor(s,  mask, 32);
            sq += __shfl_xor(sq, mask, 32);
        }
        float mu  = s * (1.f / D_DIM);
        float var = sq * (1.f / D_DIM) - mu * mu;
        float inv = rsqrtf(var + EPS);
#pragma unroll
        for (int i = 0; i < 32; ++i) {
            int d = lane + 32 * i;
            float yv = Y[r][d];
            out[(size_t)(m0 + r) * D_DIM + d] = (yv - mu) * inv * gamma[d] + beta[d];
        }
    }
}

// ---------------------------------------------------------------------------
extern "C" void kernel_launch(void* const* d_in, const int* in_sizes, int n_in,
                              void* d_out, int out_size, void* d_ws, size_t ws_size,
                              hipStream_t stream) {
    const float* x     = (const float*)d_in[0];
    const float* Wq    = (const float*)d_in[1];
    const float* bq    = (const float*)d_in[2];
    const float* Wk    = (const float*)d_in[3];
    const float* bk    = (const float*)d_in[4];
    const float* Wv    = (const float*)d_in[5];
    const float* bv    = (const float*)d_in[6];
    const float* Wo    = (const float*)d_in[7];
    const float* bo    = (const float*)d_in[8];
    const float* gamma = (const float*)d_in[9];
    const float* beta  = (const float*)d_in[10];
    float* out = (float*)d_out;

    const int M = 4 * S_LEN;                 // 8192 rows
    const int NE_X = M * D_DIM;              // 8M elems
    const int NE_W = D_DIM * D_DIM;          // 1M elems

    // workspace layout (f16 elements): 88 MB total
    _Float16* x_h   = (_Float16*)d_ws;       // 8M
    _Float16* WqT   = x_h  + (size_t)NE_X;   // 1M each
    _Float16* WkT   = WqT  + NE_W;
    _Float16* WvT   = WkT  + NE_W;
    _Float16* WoT   = WvT  + NE_W;
    _Float16* q_ws  = WoT  + NE_W;           // 8M
    _Float16* k_ws  = q_ws + (size_t)NE_X;
    _Float16* vT_ws = k_ws + (size_t)NE_X;
    _Float16* c_ws  = vT_ws + (size_t)NE_X;

    pack_f16_kernel<<<(NE_X + 255) / 256, 256, 0, stream>>>(x, x_h, NE_X);
    transpose_pack_kernel<<<(NE_W + 255) / 256, 256, 0, stream>>>(Wq, WqT, D_DIM, D_DIM);
    transpose_pack_kernel<<<(NE_W + 255) / 256, 256, 0, stream>>>(Wk, WkT, D_DIM, D_DIM);
    transpose_pack_kernel<<<(NE_W + 255) / 256, 256, 0, stream>>>(Wv, WvT, D_DIM, D_DIM);
    transpose_pack_kernel<<<(NE_W + 255) / 256, 256, 0, stream>>>(Wo, WoT, D_DIM, D_DIM);

    dim3 ggrid(M / 16, D_DIM / 512);
    qkv_gemm_kernel<<<ggrid, 256, 0, stream>>>(x_h, WqT, bq, q_ws, 0.125f, 0);
    qkv_gemm_kernel<<<ggrid, 256, 0, stream>>>(x_h, WkT, bk, k_ws, 1.0f, 0);
    qkv_gemm_kernel<<<ggrid, 256, 0, stream>>>(x_h, WvT, bv, vT_ws, 1.0f, 1);

    attn_kernel<<<dim3(S_LEN / 16, 4 * H_NUM), 32, 0, stream>>>(q_ws, k_ws, vT_ws, c_ws);

    out_proj_ln_kernel<<<M / 16, 256, 0, stream>>>(c_ws, WoT, bo, x, gamma, beta, out);
}
